// Block_24747601560224
// MI455X (gfx1250) — matrix operand order repaired and checked by reading, not compiled, not run on hardware
//
#include <hip/hip_runtime.h>
#include <hip/hip_bf16.h>

// CDNA5 (gfx1250, wave32) implementation of the Wasserstein-attention block.
// All large matmuls run on v_wmma_f32_16x16x32_f16 (f16 in, f32 accumulate).
// Fragment layouts used here (contiguous k order, not the ISA table order):
//   A (16x32 f16): lanes 0-15 -> M=lane, K=0..15 contiguous; lanes 16-31 -> K=16..31
//   B (32x16 f16): lanes 0-15 -> N=lane, K=0..15 contiguous; lanes 16-31 -> K=16..31
//   C/D (16x16 f32): VGPR v, lane l -> M = v + (l>=16 ? 8 : 0), N = l&15

typedef _Float16 f16;
typedef __attribute__((ext_vector_type(8)))  _Float16 v8h;
typedef __attribute__((ext_vector_type(16))) _Float16 v16h;
typedef __attribute__((ext_vector_type(8)))  float    v8f;

namespace cfg {
constexpr int B = 8, N = 577, D = 768, H = 12, HD = 64;
constexpr int M  = B * N;        // 4616 tokens
constexpr int MP = 4672;         // tokens padded to multiple of 64
constexpr int NP = 640;          // per-head sequence length padded (multiple of 128)
constexpr int D3 = 3 * D;        // 2304
constexpr int DM = 4 * D;        // 3072
constexpr float SCALE = 0.125f;  // 64^-0.5

constexpr size_t AL(size_t x) { return (x + 255) & ~size_t(255); }
constexpr size_t SZ_LN16 = (size_t)MP * D * 2;
constexpr size_t SZ_QKV  = (size_t)MP * D3 * 4;
constexpr size_t SZ_HEAD = (size_t)B * H * NP * HD * 2;
constexpr size_t SZ_STAT = (size_t)B * H * NP * 4;
constexpr size_t SZ_LOG  = (size_t)H * NP * NP * 4;   // per-batch
constexpr size_t SZ_ATT  = (size_t)H * NP * NP * 2;   // per-batch
constexpr size_t SZ_PROJ = (size_t)MP * D * 4;

constexpr size_t OFF_LNM  = 0;
constexpr size_t OFF_LNC  = AL(OFF_LNM + SZ_LN16);
constexpr size_t OFF_QKVM = AL(OFF_LNC + SZ_LN16);
constexpr size_t OFF_QKVC = AL(OFF_QKVM + SZ_QKV);
constexpr size_t OFF_QH   = AL(OFF_QKVC + SZ_QKV);
constexpr size_t OFF_KH   = AL(OFF_QH  + SZ_HEAD);
constexpr size_t OFF_SCQ  = AL(OFF_KH  + SZ_HEAD);
constexpr size_t OFF_SCK  = AL(OFF_SCQ + SZ_HEAD);
constexpr size_t OFF_VT   = AL(OFF_SCK + SZ_HEAD);
constexpr size_t OFF_CVT  = AL(OFF_VT  + SZ_HEAD);
constexpr size_t OFF_RA   = AL(OFF_CVT + SZ_HEAD);
constexpr size_t OFF_RB   = AL(OFF_RA  + SZ_STAT);
constexpr size_t OFF_LOG  = AL(OFF_RB  + SZ_STAT);
constexpr size_t OFF_ATT  = AL(OFF_LOG + SZ_LOG);
constexpr size_t OFF_ATT2 = AL(OFF_ATT + SZ_ATT);
constexpr size_t OFF_CTXM = AL(OFF_ATT2 + SZ_ATT);
constexpr size_t OFF_CTXC = AL(OFF_CTXM + SZ_LN16);
constexpr size_t OFF_PRJM = AL(OFF_CTXC + SZ_LN16);
constexpr size_t OFF_PRJC = AL(OFF_PRJM + SZ_PROJ);
constexpr size_t OFF_WQKV = AL(OFF_PRJC + SZ_PROJ);
constexpr size_t OFF_WPRJ = AL(OFF_WQKV + (size_t)D3 * D * 2);
constexpr size_t OFF_WCPR = AL(OFF_WPRJ + (size_t)D * D * 2);
constexpr size_t OFF_WFC1 = AL(OFF_WCPR + (size_t)D * D * 2);
constexpr size_t OFF_WFC2 = AL(OFF_WFC1 + (size_t)DM * D * 2);
constexpr size_t WS_TOTAL = AL(OFF_WFC2 + (size_t)D * DM * 2);  // ~246 MB

// Reused (dead) regions for the MLP phase:
constexpr size_t OFF_LN2 = OFF_LNM;   // layernorm-2 f16
constexpr size_t OFF_FC1 = OFF_QKVM;  // fc1 f32 output (spans qkvm+qkvc)
constexpr size_t OFF_H16 = OFF_LOG;   // gelu f16 (spans logits+attn+attn2)
constexpr size_t OFF_FC2 = OFF_QH;    // fc2 f32 output (spans head arrays)
}

static_assert(cfg::OFF_FC1 + (size_t)cfg::MP * cfg::DM * 4 <= cfg::OFF_QH,   "fc1 reuse overflow");
static_assert(cfg::OFF_H16 + (size_t)cfg::MP * cfg::DM * 2 <= cfg::OFF_CTXM, "h16 reuse overflow");
static_assert(cfg::OFF_FC2 + (size_t)cfg::MP * cfg::D  * 4 <= cfg::OFF_VT,   "fc2 reuse overflow");

__device__ __forceinline__ v8f wzero() {
  v8f z;
#pragma unroll
  for (int i = 0; i < 8; ++i) z[i] = 0.0f;
  return z;
}
__device__ __forceinline__ v16h cat16(v8h a, v8h b) {
  v16h r;
#pragma unroll
  for (int i = 0; i < 8; ++i) { r[i] = a[i]; r[i + 8] = b[i]; }
  return r;
}
__device__ __forceinline__ v8f wmma_f16(v16h a, v16h b, v8f c) {
  return __builtin_amdgcn_wmma_f32_16x16x32_f16(false, a, false, b, (short)0, c, false, false);
}
__device__ __forceinline__ float elu1(float x) { return x > 0.0f ? x + 1.0f : __expf(x); }  // elu(x)+1
__device__ __forceinline__ float gelu_exact(float x) { return 0.5f * x * (1.0f + erff(x * 0.70710678118654752f)); }

// ---------------- elementwise helpers ----------------

__global__ __launch_bounds__(256) void k_f32_to_f16(const float* __restrict__ s, f16* __restrict__ d, int n) {
  int i = blockIdx.x * 256 + threadIdx.x;
  if (i < n) d[i] = (f16)s[i];
}

__global__ __launch_bounds__(256) void k_residual(const float* __restrict__ x, const float* __restrict__ p,
                                                  const float* __restrict__ pb, const float* __restrict__ g,
                                                  float* __restrict__ out, int n) {
  int i = blockIdx.x * 256 + threadIdx.x;
  if (i < n) { int c = i % cfg::D; out[i] = x[i] + g[c] * (p[i] + pb[c]); }
}

__global__ __launch_bounds__(256) void k_final_add(float* __restrict__ out, const float* __restrict__ f2,
                                                   const float* __restrict__ b2, const float* __restrict__ g,
                                                   int n) {
  int i = blockIdx.x * 256 + threadIdx.x;
  if (i < n) { int c = i % cfg::D; out[i] += g[c] * (f2[i] + b2[c]); }
}

__global__ __launch_bounds__(256) void k_gelu(const float* __restrict__ f1, const float* __restrict__ b1,
                                              f16* __restrict__ h, int n) {
  int i = blockIdx.x * 256 + threadIdx.x;
  if (i < n) { int c = i % cfg::DM; h[i] = (f16)gelu_exact(f1[i] + b1[c]); }
}

// ---------------- layernorm (D=768, one block/row) ----------------

__global__ __launch_bounds__(256) void k_layernorm(const float* __restrict__ x, const float* __restrict__ w,
                                                   const float* __restrict__ b, f16* __restrict__ out, int rows) {
  const int r = blockIdx.x, t = threadIdx.x;
  f16* orow = out + (size_t)r * cfg::D;
  if (r >= rows) { for (int i = t; i < cfg::D; i += 256) orow[i] = (f16)0.0f; return; }
  const float* xr = x + (size_t)r * cfg::D;
  float v0 = xr[t], v1 = xr[t + 256], v2 = xr[t + 512];
  __shared__ float red1[8], red2[8];
  float s = v0 + v1 + v2;
#pragma unroll
  for (int o = 16; o > 0; o >>= 1) s += __shfl_xor(s, o, 32);
  if ((t & 31) == 0) red1[t >> 5] = s;
  __syncthreads();
  float tot = 0.0f;
#pragma unroll
  for (int i = 0; i < 8; ++i) tot += red1[i];
  const float mu = tot * (1.0f / 768.0f);
  float d0 = v0 - mu, d1 = v1 - mu, d2 = v2 - mu;
  float q = d0 * d0 + d1 * d1 + d2 * d2;
#pragma unroll
  for (int o = 16; o > 0; o >>= 1) q += __shfl_xor(q, o, 32);
  if ((t & 31) == 0) red2[t >> 5] = q;
  __syncthreads();
  float vt = 0.0f;
#pragma unroll
  for (int i = 0; i < 8; ++i) vt += red2[i];
  const float rstd = rsqrtf(vt * (1.0f / 768.0f) + 1e-5f);
  orow[t]       = (f16)(d0 * rstd * w[t]       + b[t]);
  orow[t + 256] = (f16)(d1 * rstd * w[t + 256] + b[t + 256]);
  orow[t + 512] = (f16)(d2 * rstd * w[t + 512] + b[t + 512]);
}

// ---------------- generic TN GEMM: C[MP,Nn] = A16[MP,K] * W16[Nn,K]^T ----------------
// Block = 256 threads = 8 waves (2 in M x 4 in N). Wave tile 32x64 (two A fragments
// share every B fragment: 12 b128 loads per 8 WMMAs), block tile 64x256.

__global__ __launch_bounds__(256) void k_gemm_tn(const f16* __restrict__ A, const f16* __restrict__ W,
                                                 float* __restrict__ C, int Nn, int K) {
  const int lane = threadIdx.x & 31, wid = threadIdx.x >> 5;
  const int l15 = lane & 15, lh = lane >> 4;
  const int m0 = blockIdx.y * 64 + (wid & 1) * 32;
  const int n0 = blockIdx.x * 256 + (wid >> 1) * 64;
  const int aO0 = lh ? 16 : 0, aO1 = lh ? 24 : 8, bO = lh ? 16 : 0;
  const f16* aPtr0 = A + (size_t)(m0 + l15) * K;
  const f16* aPtr1 = A + (size_t)(m0 + 16 + l15) * K;
  v8f acc[2][4];
#pragma unroll
  for (int r = 0; r < 2; ++r)
#pragma unroll
    for (int j = 0; j < 4; ++j) acc[r][j] = wzero();
  for (int k = 0; k < K; k += 32) {
    __builtin_prefetch(aPtr0 + k + 64, 0, 0);  // global_prefetch_b8
    v16h a0 = cat16(*(const v8h*)(aPtr0 + k + aO0), *(const v8h*)(aPtr0 + k + aO1));
    v16h a1 = cat16(*(const v8h*)(aPtr1 + k + aO0), *(const v8h*)(aPtr1 + k + aO1));
#pragma unroll
    for (int j = 0; j < 4; ++j) {
      const f16* bp = W + (size_t)(n0 + j * 16 + l15) * K + (k + bO);
      v16h bf = cat16(*(const v8h*)bp, *(const v8h*)(bp + 8));
      acc[0][j] = wmma_f16(a0, bf, acc[0][j]);
      acc[1][j] = wmma_f16(a1, bf, acc[1][j]);
    }
  }
#pragma unroll
  for (int r = 0; r < 2; ++r) {
#pragma unroll
    for (int j = 0; j < 4; ++j) {
      const int col = n0 + j * 16 + l15;
#pragma unroll
      for (int v = 0; v < 8; ++v) {
        const int row = m0 + r * 16 + v + (lh << 3);
        C[(size_t)row * Nn + col] = acc[r][j][v];
      }
    }
  }
}

// ---------------- head prep: split/scale/elu/sqrt + row stats + V transpose ----------------

__global__ __launch_bounds__(256) void k_prep(const float* __restrict__ qm, const float* __restrict__ qc,
                                              f16* __restrict__ qh, f16* __restrict__ kh,
                                              f16* __restrict__ scq, f16* __restrict__ sck,
                                              f16* __restrict__ vT, f16* __restrict__ cvT,
                                              float* __restrict__ rowA, float* __restrict__ rowB) {
  const int lane = threadIdx.x & 31, wid = threadIdx.x >> 5;
  const int n = blockIdx.x * 8 + wid;
  const int h = blockIdx.y, b = blockIdx.z;
  const int bh = b * cfg::H + h;
  const size_t hb = ((size_t)bh * cfg::NP + n) * 64;
  const size_t vb = (size_t)bh * 64 * cfg::NP + n;
  float sA = 0.0f, sB = 0.0f;
  if (n < cfg::N) {
    const size_t rb = (size_t)(b * cfg::N + n) * cfg::D3;
#pragma unroll
    for (int i = 0; i < 2; ++i) {
      const int d = lane + i * 32;
      const float q  = qm[rb + h * 64 + d] * cfg::SCALE;
      const float kk = qm[rb + cfg::D + h * 64 + d];
      const float vv = qm[rb + 2 * cfg::D + h * 64 + d];
      const float cq = elu1(qc[rb + h * 64 + d]);
      const float ck = elu1(qc[rb + cfg::D + h * 64 + d]);
      const float cv = elu1(qc[rb + 2 * cfg::D + h * 64 + d]);
      qh[hb + d]  = (f16)q;
      kh[hb + d]  = (f16)kk;
      scq[hb + d] = (f16)sqrtf(fmaxf(cq, 1e-24f));
      sck[hb + d] = (f16)sqrtf(fmaxf(ck, 1e-24f));
      vT[vb + (size_t)d * cfg::NP]  = (f16)vv;
      cvT[vb + (size_t)d * cfg::NP] = (f16)cv;
      sA += q * q + cq;
      sB += kk * kk + ck;
    }
  } else {
#pragma unroll
    for (int i = 0; i < 2; ++i) {
      const int d = lane + i * 32;
      qh[hb + d] = (f16)0.0f; kh[hb + d] = (f16)0.0f;
      scq[hb + d] = (f16)0.0f; sck[hb + d] = (f16)0.0f;
      vT[vb + (size_t)d * cfg::NP] = (f16)0.0f;
      cvT[vb + (size_t)d * cfg::NP] = (f16)0.0f;
    }
  }
#pragma unroll
  for (int o = 16; o > 0; o >>= 1) { sA += __shfl_xor(sA, o, 32); sB += __shfl_xor(sB, o, 32); }
  if (lane == 0) { rowA[(size_t)bh * cfg::NP + n] = sA; rowB[(size_t)bh * cfg::NP + n] = sB; }
}

// ---------------- attention logits: 2*(q.k + sqcq.sqck) - rowA[n] - rowB[m]; sigmoid; +bias ----

__global__ __launch_bounds__(256) void k_logits(const f16* __restrict__ qh, const f16* __restrict__ kh,
                                                const f16* __restrict__ scq, const f16* __restrict__ sck,
                                                const float* __restrict__ rowA, const float* __restrict__ rowB,
                                                const float* __restrict__ rpb, float* __restrict__ logits, int b) {
  const int lane = threadIdx.x & 31, wid = threadIdx.x >> 5;
  const int l15 = lane & 15, lh = lane >> 4;
  const int h = blockIdx.z;
  const int bh = b * cfg::H + h;
  const int m0 = blockIdx.x * 64;
  const int n0 = blockIdx.y * 128 + wid * 16;
  const int aO0 = lh ? 16 : 0, aO1 = lh ? 24 : 8, bO = lh ? 16 : 0;
  const f16* qp = qh  + ((size_t)bh * cfg::NP + n0 + l15) * 64;
  const f16* cp = scq + ((size_t)bh * cfg::NP + n0 + l15) * 64;
  v8f aq[4], ac[4];
#pragma unroll
  for (int j = 0; j < 4; ++j) { aq[j] = wzero(); ac[j] = wzero(); }
#pragma unroll
  for (int k = 0; k < 64; k += 32) {
    v16h afq = cat16(*(const v8h*)(qp + k + aO0), *(const v8h*)(qp + k + aO1));
    v16h afc = cat16(*(const v8h*)(cp + k + aO0), *(const v8h*)(cp + k + aO1));
#pragma unroll
    for (int j = 0; j < 4; ++j) {
      const f16* kp = kh  + ((size_t)bh * cfg::NP + m0 + j * 16 + l15) * 64 + k + bO;
      const f16* sp = sck + ((size_t)bh * cfg::NP + m0 + j * 16 + l15) * 64 + k + bO;
      v16h bk = cat16(*(const v8h*)kp, *(const v8h*)(kp + 8));
      v16h bc = cat16(*(const v8h*)sp, *(const v8h*)(sp + 8));
      aq[j] = wmma_f16(afq, bk, aq[j]);
      ac[j] = wmma_f16(afc, bc, ac[j]);
    }
  }
  float ra[8];
#pragma unroll
  for (int v = 0; v < 8; ++v) ra[v] = rowA[(size_t)bh * cfg::NP + n0 + v + (lh << 3)];
#pragma unroll
  for (int j = 0; j < 4; ++j) {
    const int m = m0 + j * 16 + l15;
    const float rb = rowB[(size_t)bh * cfg::NP + m];
#pragma unroll
    for (int v = 0; v < 8; ++v) {
      const int n = n0 + v + (lh << 3);
      const float val = 2.0f * (aq[j][v] + ac[j][v]) - ra[v] - rb;      // = -wasserstein
      const float sig = 1.0f / (1.0f + __expf(-(val + 1e-24f)));
      const float bias = (n < cfg::N && m < cfg::N) ? rpb[((size_t)h * cfg::N + n) * cfg::N + m] : 0.0f;
      logits[((size_t)h * cfg::NP + n) * cfg::NP + m] = sig + bias;
    }
  }
}

// ---------------- softmax over m (one block per (h,n) row) ----------------

__global__ __launch_bounds__(256) void k_softmax(const float* __restrict__ logits, f16* __restrict__ attn,
                                                 f16* __restrict__ attn2) {
  const int n = blockIdx.x, h = blockIdx.y, t = threadIdx.x;
  f16* ar  = attn  + ((size_t)h * cfg::NP + n) * cfg::NP;
  f16* a2r = attn2 + ((size_t)h * cfg::NP + n) * cfg::NP;
  if (n >= cfg::N) {
    for (int m = t; m < cfg::NP; m += 256) { ar[m] = (f16)0.0f; a2r[m] = (f16)0.0f; }
    return;
  }
  const float* row = logits + ((size_t)h * cfg::NP + n) * cfg::NP;
  __shared__ float red1[8], red2[8];
  float p[3];
  float mx = -1e30f;
#pragma unroll
  for (int i = 0; i < 3; ++i) { int m = t + i * 256; p[i] = (m < cfg::N) ? row[m] : -1e30f; mx = fmaxf(mx, p[i]); }
#pragma unroll
  for (int o = 16; o > 0; o >>= 1) mx = fmaxf(mx, __shfl_xor(mx, o, 32));
  if ((t & 31) == 0) red1[t >> 5] = mx;
  __syncthreads();
  float bmax = red1[0];
#pragma unroll
  for (int i = 1; i < 8; ++i) bmax = fmaxf(bmax, red1[i]);
  float s = 0.0f;
#pragma unroll
  for (int i = 0; i < 3; ++i) {
    int m = t + i * 256;
    if (m < cfg::N) { p[i] = __expf(p[i] - bmax); s += p[i]; } else p[i] = 0.0f;
  }
#pragma unroll
  for (int o = 16; o > 0; o >>= 1) s += __shfl_xor(s, o, 32);
  if ((t & 31) == 0) red2[t >> 5] = s;
  __syncthreads();
  float tot = 0.0f;
#pragma unroll
  for (int i = 0; i < 8; ++i) tot += red2[i];
  const float inv = 1.0f / tot;
#pragma unroll
  for (int i = 0; i < 3; ++i) {
    int m = t + i * 256;
    if (m < cfg::NP) { float a = p[i] * inv; ar[m] = (f16)a; a2r[m] = (f16)(a * a); }
  }
}

// ---------------- context GEMM: ctx[n,d] = attn[n,:] . vT[d,:]  (z=0 mean, z=1 cov) -------

__global__ __launch_bounds__(256) void k_ctx(const f16* __restrict__ attn, const f16* __restrict__ attn2,
                                             const f16* __restrict__ vT, const f16* __restrict__ cvT,
                                             f16* __restrict__ dstM, f16* __restrict__ dstC, int b) {
  const int lane = threadIdx.x & 31, wid = threadIdx.x >> 5;
  const int l15 = lane & 15, lh = lane >> 4;
  const int h = blockIdx.y, sel = blockIdx.z;
  const int n0 = blockIdx.x * 128 + wid * 16;
  const f16* A  = (sel ? attn2 : attn) + (size_t)h * cfg::NP * cfg::NP;
  const f16* Bm = (sel ? cvT : vT) + (size_t)(b * cfg::H + h) * 64 * cfg::NP;
  f16* dst = sel ? dstC : dstM;
  const int aO0 = lh ? 16 : 0, aO1 = lh ? 24 : 8, bO = lh ? 16 : 0;
  const f16* ap = A + (size_t)(n0 + l15) * cfg::NP;
  v8f acc[4];
#pragma unroll
  for (int j = 0; j < 4; ++j) acc[j] = wzero();
  for (int k = 0; k < cfg::NP; k += 32) {
    v16h af = cat16(*(const v8h*)(ap + k + aO0), *(const v8h*)(ap + k + aO1));
#pragma unroll
    for (int j = 0; j < 4; ++j) {
      const f16* bp = Bm + (size_t)(j * 16 + l15) * cfg::NP + k + bO;
      v16h bf = cat16(*(const v8h*)bp, *(const v8h*)(bp + 8));
      acc[j] = wmma_f16(af, bf, acc[j]);
    }
  }
#pragma unroll
  for (int j = 0; j < 4; ++j) {
    const int d = j * 16 + l15;
#pragma unroll
    for (int v = 0; v < 8; ++v) {
      const int n = n0 + v + (lh << 3);
      if (n < cfg::N) dst[(size_t)(b * cfg::N + n) * cfg::D + h * 64 + d] = (f16)acc[j][v];
    }
  }
}

// ---------------- host-side orchestration ----------------

extern "C" void kernel_launch(void* const* d_in, const int* in_sizes, int n_in,
                              void* d_out, int out_size, void* d_ws, size_t ws_size,
                              hipStream_t stream) {
  using namespace cfg;
  (void)in_sizes; (void)n_in; (void)out_size; (void)ws_size;

  const float* x_mean  = (const float*)d_in[0];
  const float* x_cov   = (const float*)d_in[1];
  const float* rpb     = (const float*)d_in[2];
  const float* n1w     = (const float*)d_in[3];
  const float* n1b     = (const float*)d_in[4];
  const float* qkv_w   = (const float*)d_in[5];
  const float* proj_w  = (const float*)d_in[6];
  const float* proj_b  = (const float*)d_in[7];
  const float* cproj_w = (const float*)d_in[8];
  const float* cproj_b = (const float*)d_in[9];
  const float* n2w     = (const float*)d_in[10];
  const float* n2b     = (const float*)d_in[11];
  const float* fc1_w   = (const float*)d_in[12];
  const float* fc1_b   = (const float*)d_in[13];
  const float* fc2_w   = (const float*)d_in[14];
  const float* fc2_b   = (const float*)d_in[15];
  const float* g1      = (const float*)d_in[16];
  const float* g2      = (const float*)d_in[17];

  char* ws = (char*)d_ws;
  f16*   lnm16  = (f16*)(ws + OFF_LNM);
  f16*   lnc16  = (f16*)(ws + OFF_LNC);
  float* qkvm   = (float*)(ws + OFF_QKVM);
  float* qkvc   = (float*)(ws + OFF_QKVC);
  f16*   qh     = (f16*)(ws + OFF_QH);
  f16*   kh     = (f16*)(ws + OFF_KH);
  f16*   scq    = (f16*)(ws + OFF_SCQ);
  f16*   sck    = (f16*)(ws + OFF_SCK);
  f16*   vT     = (f16*)(ws + OFF_VT);
  f16*   cvT    = (f16*)(ws + OFF_CVT);
  float* rowA   = (float*)(ws + OFF_RA);
  float* rowB   = (float*)(ws + OFF_RB);
  float* logits = (float*)(ws + OFF_LOG);
  f16*   attn   = (f16*)(ws + OFF_ATT);
  f16*   attn2  = (f16*)(ws + OFF_ATT2);
  f16*   ctxm   = (f16*)(ws + OFF_CTXM);
  f16*   ctxc   = (f16*)(ws + OFF_CTXC);
  float* projm  = (float*)(ws + OFF_PRJM);
  float* projc  = (float*)(ws + OFF_PRJC);
  f16*   wqkv16 = (f16*)(ws + OFF_WQKV);
  f16*   wprj16 = (f16*)(ws + OFF_WPRJ);
  f16*   wcpr16 = (f16*)(ws + OFF_WCPR);
  f16*   wfc116 = (f16*)(ws + OFF_WFC1);
  f16*   wfc216 = (f16*)(ws + OFF_WFC2);
  f16*   ln2_16 = (f16*)(ws + OFF_LN2);
  float* fc1out = (float*)(ws + OFF_FC1);
  f16*   h16    = (f16*)(ws + OFF_H16);
  float* fc2out = (float*)(ws + OFF_FC2);

  float* outm = (float*)d_out;
  float* outc = outm + (size_t)M * D;

  auto blocks = [](int n) { return (n + 255) / 256; };

  // 1) weights -> f16
  k_f32_to_f16<<<blocks(D3 * D), 256, 0, stream>>>(qkv_w,   wqkv16, D3 * D);
  k_f32_to_f16<<<blocks(D * D),  256, 0, stream>>>(proj_w,  wprj16, D * D);
  k_f32_to_f16<<<blocks(D * D),  256, 0, stream>>>(cproj_w, wcpr16, D * D);
  k_f32_to_f16<<<blocks(DM * D), 256, 0, stream>>>(fc1_w,   wfc116, DM * D);
  k_f32_to_f16<<<blocks(D * DM), 256, 0, stream>>>(fc2_w,   wfc216, D * DM);

  // 2) layernorm-1 (both streams share norm1 params)
  k_layernorm<<<MP, 256, 0, stream>>>(x_mean, n1w, n1b, lnm16, M);
  k_layernorm<<<MP, 256, 0, stream>>>(x_cov,  n1w, n1b, lnc16, M);

  // 3) QKV GEMMs
  {
    dim3 g(D3 / 256, MP / 64);
    k_gemm_tn<<<g, 256, 0, stream>>>(lnm16, wqkv16, qkvm, D3, D);
    k_gemm_tn<<<g, 256, 0, stream>>>(lnc16, wqkv16, qkvc, D3, D);
  }

  // 4) head prep (scale q, elu+1, sqrt(clip), V transpose, row stats)
  k_prep<<<dim3(NP / 8, H, B), 256, 0, stream>>>(qkvm, qkvc, qh, kh, scq, sck, vT, cvT, rowA, rowB);

  // 5) attention, batch-at-a-time to recycle the NxN scratch
  for (int b = 0; b < B; ++b) {
    k_logits<<<dim3(NP / 64, NP / 128, H), 256, 0, stream>>>(qh, kh, scq, sck, rowA, rowB, rpb, logits, b);
    k_softmax<<<dim3(NP, H), 256, 0, stream>>>(logits, attn, attn2);
    k_ctx<<<dim3(NP / 128, H, 2), 256, 0, stream>>>(attn, attn2, vT, cvT, ctxm, ctxc, b);
  }

  // 6) output projections
  {
    dim3 g(D / 256, MP / 64);
    k_gemm_tn<<<g, 256, 0, stream>>>(ctxm, wprj16, projm, D, D);
    k_gemm_tn<<<g, 256, 0, stream>>>(ctxc, wcpr16, projc, D, D);
  }

  // 7) residual: x1 = x + gamma1 * (proj + bias)   (written to d_out)
  k_residual<<<blocks(M * D), 256, 0, stream>>>(x_mean, projm, proj_b,  g1, outm, M * D);
  k_residual<<<blocks(M * D), 256, 0, stream>>>(x_cov,  projc, cproj_b, g1, outc, M * D);

  // 8) MLP for both streams (sequential, reusing scratch)
  for (int s = 0; s < 2; ++s) {
    float* xio = s ? outc : outm;
    k_layernorm<<<MP, 256, 0, stream>>>(xio, n2w, n2b, ln2_16, M);
    {
      dim3 g(DM / 256, MP / 64);
      k_gemm_tn<<<g, 256, 0, stream>>>(ln2_16, wfc116, fc1out, DM, D);
    }
    k_gelu<<<blocks(MP * DM), 256, 0, stream>>>(fc1out, fc1_b, h16, MP * DM);
    {
      dim3 g(D / 256, MP / 64);
      k_gemm_tn<<<g, 256, 0, stream>>>(h16, wfc216, fc2out, D, DM);
    }
    k_final_add<<<blocks(M * D), 256, 0, stream>>>(xio, fc2out, fc2_b, g2, M * D);
  }
}
